// GraphConvNet_70317204570386
// MI455X (gfx1250) — matrix operand order repaired and checked by reading, not compiled, not run on hardware
//
#include <hip/hip_runtime.h>
#include <hip/hip_bf16.h>

// ---------------------------------------------------------------------------
// GraphConvNet on MI455X (gfx1250): diffusion nconvs + 1x1 conv + BatchNorm
// Strategy: stream f32 adjacency once from HBM (~402 MB -> ~17us @ 23.3TB/s),
// convert to f16 in-register, run all matmuls on v_wmma_f32_16x16x32_f16.
// nconv uses a double-buffered LDS pipeline (1 barrier per K-chunk) so the
// HBM stream overlaps the WMMAs; 256 WGs/launch to cover the WGP array.
// ---------------------------------------------------------------------------

typedef __attribute__((ext_vector_type(16))) _Float16 v16h;
typedef __attribute__((ext_vector_type(8)))  _Float16 v8h;
typedef __attribute__((ext_vector_type(4)))  _Float16 v4h;
typedef __attribute__((ext_vector_type(8)))  float    v8f;
typedef __attribute__((ext_vector_type(4)))  float    v4f;

#define NB     8
#define VN     2048
#define CIN    32
#define COUT   64
#define CTOT   224     // (2*3+1)*32
#define ASTR   40      // padded LDS row stride (halfs): 80B, 16B aligned, conflict-free b128
#define WSTR   232     // padded W LDS row stride (halfs): 464B, 16B aligned
#define TW     64      // nconv rows per workgroup (4 waves x 16)

union FragAB { v16h v; v8h h[2]; };

// ---- convert x (f32) into channels [0,32) of the f16 concat buffer --------
__global__ __launch_bounds__(256) void k_cvt_x(const float* __restrict__ x,
                                               _Float16* __restrict__ Hd) {
    int idx = blockIdx.x * 256 + threadIdx.x;     // over 8*2048*32
    int row = idx >> 5, c = idx & 31;
    Hd[(size_t)row * CTOT + c] = (_Float16)x[idx];
}

// ---- nconv: out[n,w,l] = sum_v A[n,v,w] * H[n,v,src+l],  l in [0,32) ------
// grid: (32 w-tiles, 8 batches), block: 128 (4 waves); wave -> 16(w) x 32(l)
__global__ __launch_bounds__(128) void k_nconv(const float* __restrict__ A,
                                               const _Float16* __restrict__ H,
                                               _Float16* __restrict__ Hd,
                                               int src_off, int dst_off) {
    __shared__ _Float16 at[2][TW * ASTR];  // A^T tile: [w 0..63][v 0..31], double buffered
    __shared__ _Float16 xt[2][32 * ASTR];  // X^T tile: [c 0..31][v 0..31], double buffered

    const int tid    = threadIdx.x;
    const int n      = blockIdx.y;
    const int wbase  = blockIdx.x * TW;
    const int wave   = tid >> 5;
    const int lane   = tid & 31;
    const int m      = lane & 15;
    const int halfid = lane >> 4;
    const int hk     = halfid ? 8  : 0;   // A-frag K interleave (ISA table)
    const int kb     = halfid ? 8  : 0;   // B-frag K interleave (ISA table)
    const int wt     = tid & 15;          // A staging: 16 w-tiles x 8 v-tiles of 4x4
    const int vt     = tid >> 4;
    const int xv     = tid >> 2;          // X staging: 32 v rows x 4 chunks of 8 ch
    const int xcp    = (tid & 3) * 8;

    const _Float16* Hn = H + (size_t)n * VN * CTOT + src_off;
    const float*    Ab = A + (size_t)n * VN * VN + wbase;

    v8f acc0 = {}, acc1 = {};
    v4f a0, a1, a2, a3;   // staged A micro-tile (4 v-rows x 4 w)
    v8h xs;               // staged X chunk (8 channels of one v-row)

    auto stage_load = [&](int v0) {
        const float* g = Ab + (size_t)(v0 + vt * 4) * VN + wt * 4;
        a0 = *(const v4f*)(g);
        a1 = *(const v4f*)(g + VN);
        a2 = *(const v4f*)(g + 2 * VN);
        a3 = *(const v4f*)(g + 3 * VN);
        xs = *(const v8h*)(Hn + (size_t)(v0 + xv) * CTOT + xcp);
    };
    auto stage_store = [&](int buf) {
#pragma unroll
        for (int j = 0; j < 4; ++j) {
            v4h pk = { (_Float16)a0[j], (_Float16)a1[j],
                       (_Float16)a2[j], (_Float16)a3[j] };
            *(v4h*)(&at[buf][(wt * 4 + j) * ASTR + vt * 4]) = pk;  // ds_store_b64
        }
#pragma unroll
        for (int j = 0; j < 8; ++j)
            xt[buf][(xcp + j) * ASTR + xv] = xs[j];
    };

    stage_load(0);
    stage_store(0);
    __syncthreads();

    for (int v0 = 0; v0 < VN; v0 += 32) {
        const int  cur  = (v0 >> 5) & 1;
        const bool more = (v0 + 32) < VN;
        if (more) stage_load(v0 + 32);       // next chunk's HBM loads issue early

        FragAB af, bf0, bf1;
        const _Float16* ar  = &at[cur][(wave * 16 + m) * ASTR];
        af.h[0]  = *(const v8h*)(ar + hk);   // ds_load_b128
        af.h[1]  = *(const v8h*)(ar + 16 + hk);
        const _Float16* br0 = &xt[cur][m * ASTR + kb];
        const _Float16* br1 = &xt[cur][(m + 16) * ASTR + kb];
        bf0.h[0] = *(const v8h*)(br0);
        bf0.h[1] = *(const v8h*)(br0 + 16);
        bf1.h[0] = *(const v8h*)(br1);
        bf1.h[1] = *(const v8h*)(br1 + 16);

        acc0 = __builtin_amdgcn_wmma_f32_16x16x32_f16(false, af.v, false, bf0.v,
                                                      (short)0, acc0, false, false);
        acc1 = __builtin_amdgcn_wmma_f32_16x16x32_f16(false, af.v, false, bf1.v,
                                                      (short)0, acc1, false, false);

        if (more) stage_store(cur ^ 1);      // fill the other buffer
        __syncthreads();                      // single barrier per chunk
    }

    // D layout: lane -> (M = halfid*8 + r, N = m); write f16 into concat buf
    const int wrow = wbase + wave * 16 + halfid * 8;
    _Float16* Od = Hd + (size_t)n * VN * CTOT + dst_off;
#pragma unroll
    for (int r = 0; r < 8; ++r) {
        Od[(size_t)(wrow + r) * CTOT + m]      = (_Float16)acc0[r];
        Od[(size_t)(wrow + r) * CTOT + 16 + m] = (_Float16)acc1[r];
    }
}

// ---- 1x1 conv: h = H[16384,224] * W^T[224,64] + b; write pre-BN h to out;
//      also deterministic per-block channel sum/sumsq partials --------------
__global__ __launch_bounds__(256) void k_gemm2(const _Float16* __restrict__ H,
                                               const float* __restrict__ W,
                                               const float* __restrict__ b,
                                               float* __restrict__ out,
                                               float* __restrict__ part) {
    __shared__ _Float16 ws[COUT * WSTR];     // W as f16, [o][c] rows
    __shared__ float red[8][32][8];          // per-wave/lane partial sums

    const int tid = threadIdx.x;
    for (int i = tid; i < COUT * CTOT; i += 256) {
        int o = i / CTOT, c = i % CTOT;
        ws[o * WSTR + c] = (_Float16)W[i];
    }
    __syncthreads();

    const int wave    = tid >> 5;
    const int lane    = tid & 31;
    const int m       = lane & 15;
    const int halfid  = lane >> 4;
    const int hk      = halfid ? 8  : 0;
    const int kb      = halfid ? 8  : 0;
    const int rowbase = blockIdx.x * 128 + wave * 16;

    v8f acc[4] = {{}, {}, {}, {}};
    const _Float16* Hr = H + (size_t)(rowbase + m) * CTOT;

    for (int k0 = 0; k0 < CTOT; k0 += 32) {
        FragAB af;
        af.h[0] = *(const v8h*)(Hr + k0 + hk);        // global_load_b128
        af.h[1] = *(const v8h*)(Hr + k0 + 16 + hk);
#pragma unroll
        for (int t = 0; t < 4; ++t) {
            FragAB bf;
            const _Float16* br = ws + (t * 16 + m) * WSTR + k0 + kb;
            bf.h[0] = *(const v8h*)(br);
            bf.h[1] = *(const v8h*)(br + 16);
            acc[t] = __builtin_amdgcn_wmma_f32_16x16x32_f16(false, af.v, false, bf.v,
                                                            (short)0, acc[t], false, false);
        }
    }

#pragma unroll
    for (int t = 0; t < 4; ++t) {
        const int o = t * 16 + m;
        const float bb = b[o];
        float s1 = 0.f, s2 = 0.f;
#pragma unroll
        for (int r = 0; r < 8; ++r) {
            float val = acc[t][r] + bb;
            out[(size_t)(rowbase + halfid * 8 + r) * COUT + o] = val;
            s1 += val; s2 += val * val;
        }
        red[wave][lane][t * 2]     = s1;
        red[wave][lane][t * 2 + 1] = s2;
    }
    __syncthreads();
    if (tid < COUT) {  // fixed-order (deterministic) per-channel fold
        const int o = tid, t = o >> 4, col = o & 15;
        float s1 = 0.f, s2 = 0.f;
        for (int w = 0; w < 8; ++w)
            for (int h2 = 0; h2 < 2; ++h2) {
                s1 += red[w][col + h2 * 16][t * 2];
                s2 += red[w][col + h2 * 16][t * 2 + 1];
            }
        part[((size_t)blockIdx.x * COUT + o) * 2]     = s1;
        part[((size_t)blockIdx.x * COUT + o) * 2 + 1] = s2;
    }
}

// ---- fold 128 block partials -> per-channel BN scale/shift ----------------
__global__ void k_stats(const float* __restrict__ part,
                        const float* __restrict__ gamma,
                        const float* __restrict__ beta,
                        float* __restrict__ stats) {
    int o = threadIdx.x;  // 64 threads
    float s1 = 0.f, s2 = 0.f;
    for (int w = 0; w < 128; ++w) {
        s1 += part[(w * COUT + o) * 2];
        s2 += part[(w * COUT + o) * 2 + 1];
    }
    const float inv  = 1.0f / (float)(NB * VN);
    float mean = s1 * inv;
    float var  = s2 * inv - mean * mean;
    float sc   = rsqrtf(var + 1e-5f) * gamma[o];
    stats[o * 2]     = sc;
    stats[o * 2 + 1] = beta[o] - mean * sc;
}

// ---- in-place normalize ----------------------------------------------------
__global__ __launch_bounds__(256) void k_bn(float* __restrict__ out,
                                            const float* __restrict__ stats) {
    int idx = blockIdx.x * 256 + threadIdx.x;    // over 16384*64
    int o = idx & 63;
    out[idx] = out[idx] * stats[o * 2] + stats[o * 2 + 1];
}

extern "C" void kernel_launch(void* const* d_in, const int* in_sizes, int n_in,
                              void* d_out, int out_size, void* d_ws, size_t ws_size,
                              hipStream_t stream) {
    const float* x     = (const float*)d_in[0];
    const float* A0    = (const float*)d_in[1];
    const float* A1    = (const float*)d_in[2];
    const float* A2    = (const float*)d_in[3];
    const float* W     = (const float*)d_in[4];
    const float* bias  = (const float*)d_in[5];
    const float* gamma = (const float*)d_in[6];
    const float* beta  = (const float*)d_in[7];
    float* out = (float*)d_out;

    _Float16* Hbuf = (_Float16*)d_ws;                              // [8,2048,224] f16
    float* part  = (float*)((char*)d_ws + (size_t)NB * VN * CTOT * 2);
    float* stats = part + 128 * COUT * 2;

    k_cvt_x<<<(NB * VN * CIN) / 256, 256, 0, stream>>>(x, Hbuf);

    dim3 g(VN / TW, NB);   // 32 x 8 = 256 workgroups per launch
    // concat order: [x | A0x A0^2x | A1x A1^2x | A2x A2^2x]
    // per-support ordering keeps hop-2's A-stack (128 MB) resident in L2
    k_nconv<<<g, 128, 0, stream>>>(A0, Hbuf, Hbuf, 0,   32);
    k_nconv<<<g, 128, 0, stream>>>(A0, Hbuf, Hbuf, 32,  64);
    k_nconv<<<g, 128, 0, stream>>>(A1, Hbuf, Hbuf, 0,   96);
    k_nconv<<<g, 128, 0, stream>>>(A1, Hbuf, Hbuf, 96,  128);
    k_nconv<<<g, 128, 0, stream>>>(A2, Hbuf, Hbuf, 0,   160);
    k_nconv<<<g, 128, 0, stream>>>(A2, Hbuf, Hbuf, 160, 192);

    k_gemm2<<<(NB * VN) / 128, 256, 0, stream>>>(Hbuf, W, bias, out, part);
    k_stats<<<1, 64, 0, stream>>>(part, gamma, beta, stats);
    k_bn<<<(NB * VN * COUT) / 256, 256, 0, stream>>>(out, stats);
}
